// GATPrim_88596585382216
// MI455X (gfx1250) — matrix operand order repaired and checked by reading, not compiled, not run on hardware
//
#include <hip/hip_runtime.h>

// ---------------------------------------------------------------------------
// Types for CDNA5 WMMA (wave32, 16x16x32 bf16 -> f32)
// ---------------------------------------------------------------------------
typedef __bf16 bf16_t;
typedef __attribute__((ext_vector_type(16))) __bf16 v16bf;
typedef __attribute__((ext_vector_type(8)))  __bf16 v8bf;
typedef __attribute__((ext_vector_type(8)))  float  v8f;
typedef __attribute__((ext_vector_type(4)))  float  v4f;
typedef __attribute__((ext_vector_type(4)))  int    v4i;

__device__ __forceinline__ v8f wmma_bf16(v16bf a, v16bf b, v8f c) {
    // emits v_wmma_f32_16x16x32_bf16
    return __builtin_amdgcn_wmma_f32_16x16x32_bf16(
        /*neg_a=*/false, a, /*neg_b=*/false, b,
        /*c_mod=*/(short)0, c, /*reuse_a=*/false, /*reuse_b=*/false);
}

// CDNA5 async global->LDS copy (16B per lane) + completion wait.
// Issued as inline asm per the bridge doc recommendation (portable across
// toolchains, bypasses clang builtin arity differences).
__device__ __forceinline__ void async_copy_b128(const void* gptr, void* lptr) {
    unsigned long long g64 = (unsigned long long)gptr;
    unsigned l32 = (unsigned)(size_t)lptr;   // LDS byte offset = low 32 bits
    asm volatile("global_load_async_to_lds_b128 %0, %1, off"
                 :: "v"(l32), "v"(g64) : "memory");
}
__device__ __forceinline__ void async_wait0() {
    asm volatile("s_wait_asynccnt 0" ::: "memory");
}

#define ALPHA 0.2f
#define NEGV  -9.0e15f

// ---------------------------------------------------------------------------
// Elementwise f32 -> bf16
// ---------------------------------------------------------------------------
__global__ void k_f32_to_bf16(const float* __restrict__ in, bf16_t* __restrict__ out, size_t n) {
    size_t i = (size_t)blockIdx.x * blockDim.x + threadIdx.x;
    if (i < n) out[i] = (bf16_t)in[i];
}

// in: [B][R][C] f32 ; out: [B][C][R] bf16   (W -> W^T so WMMA B-fragment loads are contiguous)
__global__ void k_transpose_bf16(const float* __restrict__ in, bf16_t* __restrict__ out,
                                 int R, int C, size_t total) {
    size_t i = (size_t)blockIdx.x * blockDim.x + threadIdx.x;
    if (i >= total) return;
    size_t rc = (size_t)R * C;
    size_t b = i / rc;
    size_t rem = i - b * rc;
    int r = (int)(rem / C);
    int c = (int)(rem - (size_t)r * C);
    out[b * rc + (size_t)c * R + r] = (bf16_t)in[i];
}

// ---------------------------------------------------------------------------
// GEMM:  C[M][FOUT] = A[M][K](bf16, row-major) * Bt[FOUT][K](bf16)  per head.
// One wave computes a 16 x FOUT tile.  Block = 4 waves = 64 rows.
// Stores f32 row-major (for score dot) and bf16 transposed (for attn B side).
// ---------------------------------------------------------------------------
template<int FOUT>
__global__ void k_gemm(const bf16_t* __restrict__ A,  size_t aHeadStride,
                       const bf16_t* __restrict__ Bt, size_t bHeadStride,
                       float*  __restrict__ Crm, size_t cHeadStride,
                       bf16_t* __restrict__ Ct,  size_t ctHeadStride,
                       int M, int K) {
    constexpr int NT = FOUT / 16;
    const int lane = threadIdx.x & 31;
    const int wave = threadIdx.x >> 5;
    const int head = blockIdx.y;
    const int i0   = (blockIdx.x * 4 + wave) * 16;
    const int r    = lane & 15;
    const int half = lane >> 4;
    const int base = half * 8;

    const bf16_t* Ah  = A  + (size_t)head * aHeadStride;
    const bf16_t* Bh  = Bt + (size_t)head * bHeadStride;
    float*        Ch  = Crm + (size_t)head * cHeadStride;
    bf16_t*       Cth = Ct  + (size_t)head * ctHeadStride;

    v8f acc[NT];
#pragma unroll
    for (int t = 0; t < NT; ++t) acc[t] = (v8f)0.0f;

    union V16 { v16bf v; v8bf h[2]; };

    for (int k0 = 0; k0 < K; k0 += 32) {
        V16 a;
        const bf16_t* ap = Ah + (size_t)(i0 + r) * K + k0 + base;
        a.h[0] = *(const v8bf*)(ap);        // K = base .. base+7
        a.h[1] = *(const v8bf*)(ap + 16);   // K = base+16 .. base+23
#pragma unroll
        for (int t = 0; t < NT; ++t) {
            const bf16_t* bp = Bh + (size_t)(t * 16 + r) * K + k0 + base;
            V16 bu;
            bu.h[0] = *(const v8bf*)(bp);        // K = base .. base+7
            bu.h[1] = *(const v8bf*)(bp + 16);   // K = base+16 .. base+23
            v16bf b = bu.v;
            acc[t] = wmma_bf16(a.v, b, acc[t]);
        }
    }

    const int hi8 = half * 8;
#pragma unroll
    for (int t = 0; t < NT; ++t) {
#pragma unroll
        for (int rr = 0; rr < 8; ++rr) {
            int row = i0 + rr + hi8;
            int col = t * 16 + r;
            float v = acc[t][rr];
            Ch[(size_t)row * FOUT + col] = v;
            Cth[(size_t)col * M + row]   = (bf16_t)v;   // transposed bf16 copy
        }
    }
}

// ---------------------------------------------------------------------------
// Attention scores:  s_src = h @ a[:F],  s_dst = h @ a[F:]   (one thread per row)
// ---------------------------------------------------------------------------
template<int F>
__global__ void k_score(const float* __restrict__ h, const float* __restrict__ a,
                        float* __restrict__ s_src, float* __restrict__ s_dst, int N) {
    int idx  = blockIdx.x * blockDim.x + threadIdx.x;   // = head*N + i
    int head = idx / N;
    const float* hp = h + (size_t)idx * F;
    const float* ap = a + (size_t)head * 2 * F;
    float s1 = 0.f, s2 = 0.f;
#pragma unroll 8
    for (int f = 0; f < F; ++f) {
        float v = hp[f];
        s1 += v * ap[f];
        s2 += v * ap[F + f];
    }
    s_src[idx] = s1;
    s_dst[idx] = s2;
}

// ---------------------------------------------------------------------------
// Softmax stats: one sweep computes row max m and 1/sum for ALL heads.
// Layer 1 (WRITE_PACK): reads raw adj (the single 64MB HBM sweep) and
// ballot-packs it into a 2MB bitmask for all later kernels.
// Layer 2 (READ_PACK): reads the bitmask instead of adj (32x less traffic).
// ---------------------------------------------------------------------------
template<int H, bool WRITE_PACK, bool READ_PACK>
__global__ void k_softmax_stats(const int* __restrict__ adj,
                                const unsigned* __restrict__ packIn,   // [N][N/32]
                                unsigned* __restrict__ packOut,        // [N][N/32]
                                const float* __restrict__ s_src,       // [H][N]
                                const float* __restrict__ s_dst,       // [H][N]
                                float* __restrict__ mOut,              // [H][N]
                                float* __restrict__ invlOut,           // [H][N]
                                int N) {
    const int lane = threadIdx.x & 31;
    const int wave = threadIdx.x >> 5;
    const int i    = blockIdx.x * 4 + wave;
    const int NW   = N >> 5;

    float ssrc[H], m[H], l[H];
#pragma unroll
    for (int h = 0; h < H; ++h) {
        ssrc[h] = s_src[(size_t)h * N + i];
        m[h] = -3.0e38f;
        l[h] = 0.f;
    }
    const int* arow = adj + (size_t)i * N;
    for (int t = 0; t < NW; ++t) {
        int j = t * 32 + lane;
        int av;
        if constexpr (READ_PACK) {
            unsigned w = packIn[(size_t)i * NW + t];
            av = (w >> lane) & 1;
        } else {
            av = (arow[j] != 0);
        }
        if constexpr (WRITE_PACK) {
            unsigned w = (unsigned)__ballot(av);
            if (lane == 0) packOut[(size_t)i * NW + t] = w;
        }
#pragma unroll
        for (int h = 0; h < H; ++h) {
            float e = ssrc[h] + s_dst[(size_t)h * N + j];
            e = (e > 0.f) ? e : ALPHA * e;
            e = av ? e : NEGV;
            float mn = fmaxf(m[h], e);
            l[h] = l[h] * __expf(m[h] - mn) + __expf(e - mn);
            m[h] = mn;
        }
    }
#pragma unroll
    for (int off = 16; off >= 1; off >>= 1) {
#pragma unroll
        for (int h = 0; h < H; ++h) {
            float mo = __shfl_xor(m[h], off, 32);
            float lo = __shfl_xor(l[h], off, 32);
            float mn = fmaxf(m[h], mo);
            l[h] = l[h] * __expf(m[h] - mn) + lo * __expf(mo - mn);
            m[h] = mn;
        }
    }
    if (lane == 0) {
#pragma unroll
        for (int h = 0; h < H; ++h) {
            mOut[(size_t)h * N + i]    = m[h];
            invlOut[(size_t)h * N + i] = 1.f / l[h];
        }
    }
}

// ---------------------------------------------------------------------------
// Fused masked-softmax * V WMMA kernel (flash-style, stats precomputed):
//   out[i][:] = sum_j exp(e_ij - m_i)/l_i * h[j][:]
// - adjacency comes from the packed bitmask: 1 u32 per lane per 32-col step
// - s_dst staged once per block in LDS
// - h^T tile [FOUT x 32] staged per step in double-buffered LDS via CDNA5
//   async global->LDS copies, shared by all 4 waves of the block
// - probabilities are formed directly in the WMMA A-fragment layout:
//   lane l -> row i0+(l&15), K offsets {base..base+7, base+16..base+23}
// ---------------------------------------------------------------------------
template<int FOUT, bool TO_BF16_CONCAT>
__global__ void k_attn_av(const unsigned* __restrict__ pack,  // [N][N/32]
                          const float* __restrict__ s_src,    // [H][N]
                          const float* __restrict__ s_dst,    // [H][N]
                          const float* __restrict__ m,        // [H][N]
                          const float* __restrict__ invl,     // [H][N]
                          const bf16_t* __restrict__ ht,      // [H][FOUT][N]
                          bf16_t* __restrict__ out_bf,        // [N][catW]  (layer 1)
                          float*  __restrict__ out_f,         // [N][FOUT]  (layer 2)
                          int N, int catW) {
    constexpr int NT = FOUT / 16;
    __shared__ __align__(32) bf16_t tileB[2][FOUT * 32];   // [f][j] f-major, 64B rows
    __shared__ __align__(16) float  sdLds[4096];

    const int lane = threadIdx.x & 31;
    const int wave = threadIdx.x >> 5;
    const int head = blockIdx.y;
    const int i0   = (blockIdx.x * 4 + wave) * 16;
    const int r    = lane & 15;
    const int half = lane >> 4;
    const int base = half * 8;
    const int row_a = i0 + r;
    const int NW   = N >> 5;

    const float ssrc = s_src[(size_t)head * N + row_a];
    const float mi   = m[(size_t)head * N + row_a];
    const float* sd  = s_dst + (size_t)head * N;
    const unsigned* prow = pack + (size_t)row_a * NW;
    const bf16_t* hth = ht + (size_t)head * FOUT * N;

    // Stage s_dst for this head into LDS (coalesced, once per block).
    for (int idx = threadIdx.x; idx < N; idx += 128) sdLds[idx] = sd[idx];

    // Cooperative async tile load: FOUT*32 bf16 = FOUT*4 16B chunks.
    auto load_tile = [&](int jbase, int buf) {
        constexpr int CHUNKS = FOUT * 4;
        for (int c = threadIdx.x; c < CHUNKS; c += 128) {
            int f = c >> 2, q = c & 3;
            const bf16_t* g = hth + (size_t)f * N + jbase + q * 8;
            async_copy_b128(g, &tileB[buf][f * 32 + q * 8]);
        }
    };

    load_tile(0, 0);
    async_wait0();
    __syncthreads();

    v8f acc[NT];
#pragma unroll
    for (int t = 0; t < NT; ++t) acc[t] = (v8f)0.0f;

    union PV { v16bf v; __bf16 e[16]; };
    union DVU { v4f v[4]; float s[16]; };

    for (int j0 = 0; j0 < N; j0 += 32) {
        const int cur = (j0 >> 5) & 1;
        // Prefetch next tile into the other buffer (safe: that buffer's
        // readers finished before the barrier that ended last iteration).
        if (j0 + 32 < N) load_tile(j0 + 32, cur ^ 1);

        unsigned w = prow[j0 >> 5];        // adjacency bits for these 32 cols
        DVU dv;
        dv.v[0] = *(const v4f*)(sdLds + j0 + base);
        dv.v[1] = *(const v4f*)(sdLds + j0 + base + 4);
        dv.v[2] = *(const v4f*)(sdLds + j0 + base + 16);
        dv.v[3] = *(const v4f*)(sdLds + j0 + base + 20);

        PV p;
#pragma unroll
        for (int e = 0; e < 16; ++e) {
            int bit = base + ((e < 8) ? e : (8 + e));   // base+e  /  base+16+(e-8)
            float ev = ssrc + dv.s[e];
            ev = (ev > 0.f) ? ev : ALPHA * ev;
            ev = ((w >> bit) & 1u) ? ev : NEGV;
            p.e[e] = (__bf16)__expf(ev - mi);           // masked -> 0
        }

#pragma unroll
        for (int t = 0; t < NT; ++t) {
            union { v16bf v; v8bf h[2]; } bu;
            bu.h[0] = *(const v8bf*)&tileB[cur][(t * 16 + r) * 32 + base];
            bu.h[1] = *(const v8bf*)&tileB[cur][(t * 16 + r) * 32 + base + 16];
            v16bf b = bu.v;
            acc[t] = wmma_bf16(p.v, b, acc[t]);
        }

        async_wait0();       // my async writes to the other buffer landed
        __syncthreads();     // everyone done reading cur & writes visible
    }

    const int hi8 = half * 8;
#pragma unroll
    for (int rr = 0; rr < 8; ++rr) {
        int row = i0 + rr + hi8;
        float il = invl[(size_t)head * N + row];
#pragma unroll
        for (int t = 0; t < NT; ++t) {
            float v = acc[t][rr] * il;
            v = (v > 0.f) ? v : (__expf(v) - 1.f);   // ELU
            int col = t * 16 + r;
            if constexpr (TO_BF16_CONCAT)
                out_bf[(size_t)row * catW + head * FOUT + col] = (bf16_t)v;
            else
                out_f[(size_t)row * FOUT + col] = v;
        }
    }
}

// ---------------------------------------------------------------------------
// Host launch
// ---------------------------------------------------------------------------
extern "C" void kernel_launch(void* const* d_in, const int* in_sizes, int n_in,
                              void* d_out, int out_size, void* d_ws, size_t ws_size,
                              hipStream_t stream) {
    (void)in_sizes; (void)n_in; (void)out_size; (void)ws_size;
    const float* x     = (const float*)d_in[0];   // [4096,512]
    const int*   adj   = (const int*)  d_in[1];   // [4096,4096]
    const float* W_att = (const float*)d_in[2];   // [8,512,64]
    const float* a_att = (const float*)d_in[3];   // [8,128,1]
    const float* W_out = (const float*)d_in[4];   // [512,32]
    const float* a_out = (const float*)d_in[5];   // [64,1]

    constexpr int N = 4096, FIN = 512, FH = 64, FC = 32, H = 8;

    char* p = (char*)d_ws;
    auto carve = [&](size_t bytes) -> char* {
        char* q = p;
        p += (bytes + 255) & ~(size_t)255;
        return q;
    };
    bf16_t*   xbf  = (bf16_t*)carve((size_t)N * FIN * 2);
    bf16_t*   Wt   = (bf16_t*)carve((size_t)H * FH * FIN * 2);   // [H][FH][FIN]
    bf16_t*   Wto  = (bf16_t*)carve((size_t)FC * FIN * 2);       // [FC][FIN]
    float*    hf   = (float*) carve((size_t)H * N * FH * 4);     // [H][N][FH]
    bf16_t*   htb  = (bf16_t*)carve((size_t)H * FH * N * 2);     // [H][FH][N]
    float*    ss1  = (float*) carve((size_t)H * N * 4);
    float*    sd1  = (float*) carve((size_t)H * N * 4);
    float*    m1   = (float*) carve((size_t)H * N * 4);
    float*    il1  = (float*) carve((size_t)H * N * 4);
    bf16_t*   hcat = (bf16_t*)carve((size_t)N * H * FH * 2);     // [N][512]
    float*    h2f  = (float*) carve((size_t)N * FC * 4);         // [N][32]
    bf16_t*   ht2  = (bf16_t*)carve((size_t)FC * N * 2);         // [32][N]
    float*    ss2  = (float*) carve((size_t)N * 4);
    float*    sd2  = (float*) carve((size_t)N * 4);
    float*    m2   = (float*) carve((size_t)N * 4);
    float*    il2  = (float*) carve((size_t)N * 4);
    unsigned* pck  = (unsigned*)carve((size_t)N * (N / 32) * 4); // 2MB bitmask

    // --- precision conversions / weight transposes ---
    {
        size_t n = (size_t)N * FIN;
        k_f32_to_bf16<<<dim3((unsigned)((n + 255) / 256)), 256, 0, stream>>>(x, xbf, n);
    }
    {
        size_t n = (size_t)H * FIN * FH;
        k_transpose_bf16<<<dim3((unsigned)((n + 255) / 256)), 256, 0, stream>>>(W_att, Wt, FIN, FH, n);
    }
    {
        size_t n = (size_t)FIN * FC;
        k_transpose_bf16<<<dim3((unsigned)((n + 255) / 256)), 256, 0, stream>>>(W_out, Wto, FIN, FC, n);
    }

    // --- layer 1: h = x @ W  (per head), WMMA ---
    k_gemm<FH><<<dim3(N / 64, H), 128, 0, stream>>>(
        xbf, 0, Wt, (size_t)FH * FIN, hf, (size_t)N * FH, htb, (size_t)FH * N, N, FIN);

    // --- layer 1: scores; stats sweep adj once (all 8 heads) + bitmask pack ---
    k_score<FH><<<dim3((H * N) / 256), 256, 0, stream>>>(hf, a_att, ss1, sd1, N);
    k_softmax_stats<H, true, false><<<dim3(N / 4), 128, 0, stream>>>(
        adj, nullptr, pck, ss1, sd1, m1, il1, N);

    // --- layer 1: fused softmax * V with WMMA + async-LDS, ELU, concat (bf16) ---
    k_attn_av<FH, true><<<dim3(N / 64, H), 128, 0, stream>>>(
        pck, ss1, sd1, m1, il1, htb, hcat, nullptr, N, H * FH);

    // --- layer 2: h2 = hcat @ W_out, WMMA ---
    k_gemm<FC><<<dim3(N / 64, 1), 128, 0, stream>>>(
        hcat, 0, Wto, 0, h2f, 0, ht2, 0, N, FIN);

    // --- layer 2: scores + stats (bitmask) + fused attention, ELU -> f32 out ---
    k_score<FC><<<dim3(N / 256), 256, 0, stream>>>(h2f, a_out, ss2, sd2, N);
    k_softmax_stats<1, false, true><<<dim3(N / 4), 128, 0, stream>>>(
        adj, pck, nullptr, ss2, sd2, m2, il2, N);
    k_attn_av<FC, false><<<dim3(N / 64, 1), 128, 0, stream>>>(
        pck, ss2, sd2, m2, il2, ht2, nullptr, (float*)d_out, N, FC);
}
